// DiplomacyModel_33122787787224
// MI455X (gfx1250) — matrix operand order repaired and checked by reading, not compiled, not run on hardware
//
#include <hip/hip_runtime.h>

#define NLOC 81
#define BATCHN 2048
#define EMB 256
#define NBLK 8
#define EPS 1e-5f

typedef __attribute__((ext_vector_type(16))) __bf16 v16bf;
typedef __attribute__((ext_vector_type(2)))  __bf16 v2bf;
typedef __attribute__((ext_vector_type(8)))  float  v8f;

#if defined(__has_builtin)
#  if __has_builtin(__builtin_amdgcn_cvt_pk_bf16_f32)
#    define HAVE_CVT_PK_BF16 1
#  endif
#endif

union FragU {
    uint4 q[2];
    v16bf v;
};

__device__ __forceinline__ unsigned pack2bf(float lo, float hi) {
#ifdef HAVE_CVT_PK_BF16
    union { v2bf v; unsigned u; } c;
    c.v = __builtin_amdgcn_cvt_pk_bf16_f32(lo, hi);
    return c.u;
#else
    // native fptrunc: lets the backend pick v_cvt_pk_bf16_f32 if available
    union { v2bf v; unsigned u; } c;
    c.v[0] = (__bf16)lo;
    c.v[1] = (__bf16)hi;
    return c.u;
#endif
}

__device__ __forceinline__ unsigned short f2bf(float f) {
#ifdef HAVE_CVT_PK_BF16
    union { v2bf v; unsigned short s[2]; } c;
    c.v = __builtin_amdgcn_cvt_pk_bf16_f32(f, 0.0f);
    return c.s[0];
#else
    union { __bf16 b; unsigned short s; } c;
    c.b = (__bf16)f;
    return c.s;
#endif
}

// ---------------------------------------------------------------------------
// FAST per-node GEMM for F == EMB (layers 1..7):
//   T[b][n][o] = sum_i X[b][n][i] * W[n][i][o]
// 256 threads = 8 waves; 128x128 output tile; K stepped by 32 (bf16 WMMA);
// double-buffered LDS, no bounds checks, vectorized staging.
// Wave layout: wr = wave>>1 (4 rows of 32), wc = wave&1 (2 cols of 64);
// each wave computes 2 M-tiles x 4 N-tiles of 16x16.
// ---------------------------------------------------------------------------
__global__ __launch_bounds__(256)
void k_gemm_node_fast(const float* __restrict__ X, const float* __restrict__ W,
                      float* __restrict__ T)
{
    const int n   = blockIdx.z;
    const int mb  = blockIdx.x * 128;    // batch-row base
    const int nb  = blockIdx.y * 128;    // out-col base
    const int tid = threadIdx.x;
    const int lane = tid & 31;
    const int wave = tid >> 5;
    const int wr = wave >> 1;            // 0..3
    const int wc = wave & 1;             // 0..1
    const int lm = lane & 15;
    const int half = lane >> 4;

    const float* __restrict__ Xb = X + ((size_t)mb * NLOC + n) * EMB;
    const float* __restrict__ Wn = W + (size_t)n * EMB * EMB;
    const size_t xRowStride = (size_t)NLOC * EMB;

    // double-buffered LDS: Xs row-major [128 rows][32 k] (stride 40 bf16),
    //                      Ws transposed [128 o][32 k]
    __shared__ __align__(16) unsigned short Xs[2][128 * 40];
    __shared__ __align__(16) unsigned short Ws[2][128 * 40];

    v8f acc[2][4] = {};

    // staging registers (global -> reg -> LDS pipeline)
    float4 xr4[4], wr4[4];

    // prologue: load K-step 0
    #pragma unroll
    for (int p = 0; p < 4; ++p) {
        int idx = p * 256 + tid;
        int row = idx >> 3;
        int kq  = (idx & 7) * 4;
        xr4[p] = *(const float4*)&Xb[(size_t)row * xRowStride + kq];
        int k  = idx >> 5;
        int oq = (idx & 31) * 4;
        wr4[p] = *(const float4*)&Wn[(size_t)k * EMB + nb + oq];
    }

    const int KSTEPS = EMB / 32;         // 8
    for (int ks = 0; ks < KSTEPS; ++ks) {
        const int buf = ks & 1;
        // commit staged registers to LDS (bf16)
        #pragma unroll
        for (int p = 0; p < 4; ++p) {
            int idx = p * 256 + tid;
            int row = idx >> 3;
            int kq  = (idx & 7) * 4;
            uint2 u;
            u.x = pack2bf(xr4[p].x, xr4[p].y);
            u.y = pack2bf(xr4[p].z, xr4[p].w);
            *(uint2*)&Xs[buf][row * 40 + kq] = u;
            int k  = idx >> 5;
            int oq = (idx & 31) * 4;
            Ws[buf][(oq + 0) * 40 + k] = f2bf(wr4[p].x);
            Ws[buf][(oq + 1) * 40 + k] = f2bf(wr4[p].y);
            Ws[buf][(oq + 2) * 40 + k] = f2bf(wr4[p].z);
            Ws[buf][(oq + 3) * 40 + k] = f2bf(wr4[p].w);
        }
        __syncthreads();

        // prefetch next K-step into registers (overlaps with WMMA below)
        if (ks + 1 < KSTEPS) {
            const int kk = (ks + 1) * 32;
            #pragma unroll
            for (int p = 0; p < 4; ++p) {
                int idx = p * 256 + tid;
                int row = idx >> 3;
                int kq  = (idx & 7) * 4;
                xr4[p] = *(const float4*)&Xb[(size_t)row * xRowStride + kk + kq];
                int k  = idx >> 5;
                int oq = (idx & 31) * 4;
                wr4[p] = *(const float4*)&Wn[(size_t)(kk + k) * EMB + nb + oq];
            }
        }

        // fragments (ISA 16-bit A layout: lane=row, two 8-k runs;
        //            B layout: lane=col, two 8-k runs, same as A)
        FragU fa[2], fb[4];
        #pragma unroll
        for (int mt = 0; mt < 2; ++mt) {
            int m = wr * 32 + mt * 16 + lm;
            fa[mt].q[0] = *(const uint4*)&Xs[buf][m * 40 + half * 8];
            fa[mt].q[1] = *(const uint4*)&Xs[buf][m * 40 + 16 + half * 8];
        }
        #pragma unroll
        for (int nt = 0; nt < 4; ++nt) {
            int o = wc * 64 + nt * 16 + lm;
            fb[nt].q[0] = *(const uint4*)&Ws[buf][o * 40 + half * 8];
            fb[nt].q[1] = *(const uint4*)&Ws[buf][o * 40 + 16 + half * 8];
        }
        #pragma unroll
        for (int mt = 0; mt < 2; ++mt)
            #pragma unroll
            for (int nt = 0; nt < 4; ++nt)
                acc[mt][nt] = __builtin_amdgcn_wmma_f32_16x16x32_bf16(
                    false, fa[mt].v, false, fb[nt].v,
                    (short)0, acc[mt][nt], false, false);
    }

    // write back (C/D layout: VGPR r -> M = half*8 + r, N = lane%16)
    #pragma unroll
    for (int mt = 0; mt < 2; ++mt) {
        #pragma unroll
        for (int nt = 0; nt < 4; ++nt) {
            int row0 = mb + wr * 32 + mt * 16 + half * 8;
            int col  = nb + wc * 64 + nt * 16 + lm;
            #pragma unroll
            for (int r = 0; r < 8; ++r)
                T[((size_t)(row0 + r) * NLOC + n) * EMB + col] = acc[mt][nt][r];
        }
    }
}

// ---------------------------------------------------------------------------
// Generic (guarded) per-node GEMM, used only for layer 0 (F = 35 / 40).
// 128 threads = 4 waves; 64x64 tile.
// ---------------------------------------------------------------------------
__global__ __launch_bounds__(128)
void k_gemm_node_first(const float* __restrict__ X, const float* __restrict__ W,
                       float* __restrict__ T, int F)
{
    const int n   = blockIdx.z;
    const int mb  = blockIdx.x * 64;
    const int nb  = blockIdx.y * 64;
    const int tid = threadIdx.x;
    const int lane = tid & 31;
    const int wave = tid >> 5;
    const int wr = wave >> 1;
    const int wc = wave & 1;
    const int lm = lane & 15;
    const int half = lane >> 4;

    const float* __restrict__ Wn = W + (size_t)n * F * EMB;

    __shared__ __align__(16) unsigned short Xs[64 * 40];
    __shared__ __align__(16) unsigned short Ws[64 * 40];

    v8f acc[2][2] = {};

    const int KSTEPS = (F + 31) >> 5;
    for (int ks = 0; ks < KSTEPS; ++ks) {
        const int kk = ks * 32;
        __syncthreads();
        for (int i = tid; i < 64 * 16; i += 128) {
            int row = i >> 4;
            int kp  = (i & 15) * 2;
            int gm  = mb + row;
            int gk  = kk + kp;
            const float* xr = X + ((size_t)gm * NLOC + n) * F;
            float f0 = (gk     < F) ? xr[gk]     : 0.0f;
            float f1 = (gk + 1 < F) ? xr[gk + 1] : 0.0f;
            *(unsigned*)&Xs[row * 40 + kp] = pack2bf(f0, f1);
        }
        for (int i = tid; i < 32 * 64; i += 128) {
            int k = i >> 6;
            int o = i & 63;
            int gk = kk + k;
            float f = (gk < F) ? Wn[(size_t)gk * EMB + nb + o] : 0.0f;
            Ws[o * 40 + k] = f2bf(f);
        }
        __syncthreads();

        FragU fa[2], fb[2];
        #pragma unroll
        for (int mt = 0; mt < 2; ++mt) {
            int m = wr * 32 + mt * 16 + lm;
            fa[mt].q[0] = *(const uint4*)&Xs[m * 40 + half * 8];
            fa[mt].q[1] = *(const uint4*)&Xs[m * 40 + 16 + half * 8];
        }
        #pragma unroll
        for (int nt = 0; nt < 2; ++nt) {
            int o = wc * 32 + nt * 16 + lm;
            fb[nt].q[0] = *(const uint4*)&Ws[o * 40 + half * 8];
            fb[nt].q[1] = *(const uint4*)&Ws[o * 40 + 16 + half * 8];
        }
        #pragma unroll
        for (int mt = 0; mt < 2; ++mt)
            #pragma unroll
            for (int nt = 0; nt < 2; ++nt)
                acc[mt][nt] = __builtin_amdgcn_wmma_f32_16x16x32_bf16(
                    false, fa[mt].v, false, fb[nt].v,
                    (short)0, acc[mt][nt], false, false);
    }

    #pragma unroll
    for (int mt = 0; mt < 2; ++mt) {
        #pragma unroll
        for (int nt = 0; nt < 2; ++nt) {
            int row0 = mb + wr * 32 + mt * 16 + half * 8;
            int col  = nb + wc * 32 + nt * 16 + lm;
            #pragma unroll
            for (int r = 0; r < 8; ++r)
                T[((size_t)(row0 + r) * NLOC + n) * EMB + col] = acc[mt][nt][r];
        }
    }
}

// ---------------------------------------------------------------------------
// Adjacency mix: Y[b][m][o] = bias[o] + sum_n A[m][n] * T[b][n][o]
// ---------------------------------------------------------------------------
__global__ __launch_bounds__(256)
void k_adjmix(const float* __restrict__ T, const float* __restrict__ A,
              const float* __restrict__ bias, float* __restrict__ Y)
{
    extern __shared__ float sm[];
    float* tS = sm;                 // [NLOC][EMB]
    float* aS = sm + NLOC * EMB;    // [NLOC][NLOC]
    const int b   = blockIdx.x;
    const int tid = threadIdx.x;

    const float* __restrict__ Tb = T + (size_t)b * NLOC * EMB;
    for (int i = tid; i < NLOC * EMB; i += 256) tS[i] = Tb[i];
    for (int i = tid; i < NLOC * NLOC; i += 256) aS[i] = A[i];
    __syncthreads();

    const int o = tid;
    const float bia = bias[o];
    float* __restrict__ Yb = Y + (size_t)b * NLOC * EMB;
    for (int m = 0; m < NLOC; ++m) {
        float s = bia;
        #pragma unroll 9
        for (int nn = 0; nn < NLOC; ++nn)
            s = fmaf(aS[m * NLOC + nn], tS[nn * EMB + o], s);
        Yb[(size_t)m * EMB + o] = s;
    }
}

// ---------------------------------------------------------------------------
// BatchNorm statistics per node; stats: [0..80]=sum [81..161]=sumsq
//                                       [162..242]=scale [243..323]=shift
// ---------------------------------------------------------------------------
__global__ void k_zero_stats(float* __restrict__ stats)
{
    int i = threadIdx.x;
    if (i < 4 * NLOC) stats[i] = 0.0f;
}

__global__ __launch_bounds__(256)
void k_stats(const float* __restrict__ Y, float* __restrict__ stats)
{
    const int m   = blockIdx.x;
    const int tid = threadIdx.x;
    const int b0  = blockIdx.y * 128;
    float s1 = 0.0f, s2 = 0.0f;
    #pragma unroll 4
    for (int j = 0; j < 128; ++j) {
        float v = Y[((size_t)(b0 + j) * NLOC + m) * EMB + tid];
        s1 += v;
        s2 += v * v;
    }
    __shared__ float r1[256], r2[256];
    r1[tid] = s1; r2[tid] = s2;
    __syncthreads();
    #pragma unroll
    for (int s = 128; s > 0; s >>= 1) {
        if (tid < s) { r1[tid] += r1[tid + s]; r2[tid] += r2[tid + s]; }
        __syncthreads();
    }
    if (tid == 0) {
        atomicAdd(&stats[m], r1[0]);
        atomicAdd(&stats[NLOC + m], r2[0]);
    }
}

__global__ void k_finalize(float* __restrict__ stats,
                           const float* __restrict__ g,
                           const float* __restrict__ be)
{
    int m = threadIdx.x;
    if (m < NLOC) {
        const float inv = 1.0f / ((float)BATCHN * (float)EMB);
        float mean = stats[m] * inv;
        float var  = stats[NLOC + m] * inv - mean * mean;
        float rs   = rsqrtf(var + EPS);
        float sc   = rs * g[m];
        stats[2 * NLOC + m] = sc;
        stats[3 * NLOC + m] = be[m] - mean * sc;
    }
}

// ---------------------------------------------------------------------------
// Normalize + ReLU (+ residual); strided/offset output for the final concat
// ---------------------------------------------------------------------------
__global__ __launch_bounds__(256)
void k_norm(const float* __restrict__ Y, const float* __restrict__ stats,
            const float* __restrict__ Xres, float* __restrict__ out,
            int outStride, int outOff)
{
    size_t idx = (size_t)blockIdx.x * 256 + threadIdx.x;
    int o = (int)(idx & (EMB - 1));
    size_t bm = idx >> 8;
    int m = (int)(bm % NLOC);
    size_t b = bm / NLOC;
    float sc = stats[2 * NLOC + m];
    float sh = stats[3 * NLOC + m];
    float v = fmaf(Y[idx], sc, sh);
    v = fmaxf(v, 0.0f);
    if (Xres) v += Xres[idx];
    out[(b * NLOC + m) * (size_t)outStride + outOff + o] = v;
}

// ---------------------------------------------------------------------------
// Host driver
// ---------------------------------------------------------------------------
static void run_stack(const float* x_in, int F0, const float* A,
                      const float* W0, const float* b0,
                      const float* g0, const float* be0,
                      const float* Ws, const float* bs,
                      const float* gs, const float* bes,
                      float* buf0, float* buf1, float* buf2, float* stats,
                      float* d_out, int outOff, hipStream_t stream)
{
    float* buf[3] = { buf0, buf1, buf2 };
    const size_t adjSmem = (size_t)(NLOC * EMB + NLOC * NLOC) * sizeof(float);
    const int normBlocks = (BATCHN * NLOC * EMB) / 256;

    for (int i = 0; i < NBLK; ++i) {
        const float* Xin  = (i == 0) ? x_in : buf[(i - 1) % 3];
        const float* Wp   = (i == 0) ? W0 : Ws + (size_t)(i - 1) * NLOC * EMB * EMB;
        const float* bias = (i == 0) ? b0 : bs + (size_t)(i - 1) * EMB;
        const float* g    = (i == 0) ? g0 : gs + (size_t)(i - 1) * NLOC;
        const float* be   = (i == 0) ? be0 : bes + (size_t)(i - 1) * NLOC;
        float* T  = buf[i % 3];
        float* Yb = buf[(i + 1) % 3];

        if (i == 0) {
            dim3 gg(BATCHN / 64, EMB / 64, NLOC);
            k_gemm_node_first<<<gg, 128, 0, stream>>>(Xin, Wp, T, F0);
        } else {
            dim3 gg(BATCHN / 128, EMB / 128, NLOC);
            k_gemm_node_fast<<<gg, 256, 0, stream>>>(Xin, Wp, T);
        }
        k_adjmix<<<BATCHN, 256, adjSmem, stream>>>(T, A, bias, Yb);
        k_zero_stats<<<1, 512, 0, stream>>>(stats);
        k_stats<<<dim3(NLOC, BATCHN / 128), 256, 0, stream>>>(Yb, stats);
        k_finalize<<<1, 128, 0, stream>>>(stats, g, be);

        const float* Xres = (i == 0) ? nullptr : Xin;
        if (i == NBLK - 1) {
            k_norm<<<normBlocks, 256, 0, stream>>>(Yb, stats, Xres, d_out, 2 * EMB, outOff);
        } else {
            k_norm<<<normBlocks, 256, 0, stream>>>(Yb, stats, Xres, T, EMB, 0);
        }
    }
}

extern "C" void kernel_launch(void* const* d_in, const int* in_sizes, int n_in,
                              void* d_out, int out_size, void* d_ws, size_t ws_size,
                              hipStream_t stream)
{
    const float* x_bo   = (const float*)d_in[0];
    const float* x_po   = (const float*)d_in[1];
    const float* A      = (const float*)d_in[2];
    const float* W0_bo  = (const float*)d_in[3];
    const float* b0_bo  = (const float*)d_in[4];
    const float* g0_bo  = (const float*)d_in[5];
    const float* be0_bo = (const float*)d_in[6];
    const float* W_bo   = (const float*)d_in[7];
    const float* b_bo   = (const float*)d_in[8];
    const float* g_bo   = (const float*)d_in[9];
    const float* be_bo  = (const float*)d_in[10];
    const float* W0_po  = (const float*)d_in[11];
    const float* b0_po  = (const float*)d_in[12];
    const float* g0_po  = (const float*)d_in[13];
    const float* be0_po = (const float*)d_in[14];
    const float* W_po   = (const float*)d_in[15];
    const float* b_po   = (const float*)d_in[16];
    const float* g_po   = (const float*)d_in[17];
    const float* be_po  = (const float*)d_in[18];

    char* ws = (char*)d_ws;
    const size_t bufBytes = (size_t)BATCHN * NLOC * EMB * sizeof(float);
    float* buf0  = (float*)(ws);
    float* buf1  = (float*)(ws + bufBytes);
    float* buf2  = (float*)(ws + 2 * bufBytes);
    float* stats = (float*)(ws + 3 * bufBytes);

    float* out = (float*)d_out;   // [B][81][512]

    run_stack(x_bo, 35, A, W0_bo, b0_bo, g0_bo, be0_bo,
              W_bo, b_bo, g_bo, be_bo,
              buf0, buf1, buf2, stats, out, 0, stream);
    run_stack(x_po, 40, A, W0_po, b0_po, g0_po, be0_po,
              W_po, b_po, g_po, be_po,
              buf0, buf1, buf2, stats, out, EMB, stream);
}
